// GeniePath_15917148799863
// MI455X (gfx1250) — matrix operand order repaired and checked by reading, not compiled, not run on hardware
//
#include <hip/hip_runtime.h>
#include <hip/hip_bf16.h>

// ---------------------------------------------------------------------------
// GeniePath (GAT + LSTM, 3 layers) for MI455X / gfx1250.
// Dense GEMMs via v_wmma_f32_16x16x32_bf16:
//   - B (shared weight tile) double-buffered in LDS (2 x 9 KB, rows padded to
//     72 elems for conflict-free 16-bit fragment reads)
//   - A fragments load global->register directly (no cross-wave sharing),
//     ping-ponged one K-stage ahead so loads overlap the 8-WMMA stage
//   - __launch_bounds__(256,1) to avoid occupancy-driven VGPR spills
//   - global_prefetch_b8 two stages ahead on the streaming A operand
// LSTM gate GEMMs fused into one K=512 pass: gates = [xb|h] @ [w_ih^T; w_hh^T].
// Edge softmax via f32 atomics (max via order-preserving uint encoding).
// ---------------------------------------------------------------------------

typedef __attribute__((ext_vector_type(16))) __bf16 v16bf;
typedef __attribute__((ext_vector_type(8)))  float  v8f;

#define DIMH 256
#define NEG_SLOPE 0.2f
#define RES_W 0.1f

#define BM 128
#define BN 64
#define BK 64
#define LPAD 72   // LDS row stride (ushorts): 144B -> conflict-free 16-bit reads

// ---------------- helpers ----------------
__device__ __forceinline__ unsigned short f2bf(float f) {
  unsigned u = __float_as_uint(f);
  u += 0x7FFFu + ((u >> 16) & 1u);       // round-to-nearest-even
  return (unsigned short)(u >> 16);
}
__device__ __forceinline__ unsigned f2ord(float f) {  // order-preserving uint
  unsigned u = __float_as_uint(f);
  return (u & 0x80000000u) ? ~u : (u | 0x80000000u);
}
__device__ __forceinline__ float ord2f(unsigned u) {
  return __uint_as_float((u & 0x80000000u) ? (u & 0x7FFFFFFFu) : ~u);
}
__device__ __forceinline__ float sigm(float x) { return 1.0f / (1.0f + __expf(-x)); }

// ---------------- conversion kernels ----------------
// out[row * ostride + ooff + col] = bf16( (tanh?) in[row * cols + col] )
__global__ void k_to_bf16_strided(const float* __restrict__ in,
                                  unsigned short* __restrict__ out,
                                  long total, int cols, int ostride, int ooff,
                                  int do_tanh) {
  long idx = (long)blockIdx.x * blockDim.x + threadIdx.x;
  if (idx >= total) return;
  long r = idx / cols;
  int  ccol = (int)(idx % cols);
  float v = in[idx];
  if (do_tanh) v = tanhf(v);
  out[r * ostride + ooff + ccol] = f2bf(v);
}

// Build bf16 weight [K x Npad] (zero padded cols). trans=0: in is [K x Nout].
// trans=1: in is [Nout x K].
__global__ void k_weight_to_bf16(const float* __restrict__ in,
                                 unsigned short* __restrict__ out,
                                 int K, int Nout, int Npad, int trans) {
  int idx = blockIdx.x * blockDim.x + threadIdx.x;
  if (idx >= K * Npad) return;
  int k = idx / Npad, j = idx % Npad;
  float v = 0.0f;
  if (j < Nout) v = trans ? in[(size_t)j * K + k] : in[(size_t)k * Nout + j];
  out[idx] = f2bf(v);
}

// Concatenated gate weight: out[k*G + j] = (k<D ? w_ih[j,k] : w_hh[j,k-D]),
// out is [2D x G] bf16 (the [w_ih^T ; w_hh^T] stack).
__global__ void k_gates_weight(const float* __restrict__ wih,
                               const float* __restrict__ whh,
                               unsigned short* __restrict__ out, int D, int G) {
  int idx = blockIdx.x * blockDim.x + threadIdx.x;
  if (idx >= 2 * D * G) return;
  int k = idx / G, j = idx % G;
  float v = (k < D) ? wih[(size_t)j * D + k] : whh[(size_t)j * D + (k - D)];
  out[idx] = f2bf(v);
}

// ---------------- WMMA GEMM (B in LDS, A ping-ponged in registers) --------
// C[M x Nout](f32) = A[M x K](bf16, lda=K) @ B[K x Npad](bf16) [+ bias]
// K must be a multiple of 128 or == 128 (KT even; here K in {128,256,512}).
// 256 threads = 8 waves; block tile 128(M) x 64(N); wave tile 16(M) x 64(N).
__global__ __launch_bounds__(256, 1)
void k_gemm_bf16(const unsigned short* __restrict__ A,
                 const unsigned short* __restrict__ B,
                 float* __restrict__ C, const float* __restrict__ bias,
                 int M, int K, int Nout, int Npad) {
  __shared__ unsigned short Bs[2][BK * LPAD];   // 2 x 9 KB

  const int t    = threadIdx.x;
  const int lane = t & 31;
  const int wave = t >> 5;
  const int m0b  = blockIdx.x * BM;
  const int n0b  = blockIdx.y * BN;

  // B global->reg staging: uint4 = 8 bf16; thread t covers rows sr, sr+32
  const int sr = t >> 3;          // base row 0..31
  const int sc = (t & 7) * 8;     // col (elements) 0..56
  const int KT = K / BK;

  // A fragment addressing (16-bit A 16x32 layout)
  const int m0w = wave * 16;
  const int am  = lane & 15;
  const int ak  = (lane >> 4) * 8;
  int arow = m0b + m0w + am;
  if (arow >= M) arow = M - 1;    // clamp; C stores are guarded
  const unsigned short* Arow = A + (size_t)arow * K;

  uint4 rb[2];
  auto gloadB = [&](int kt) {
    int k0 = kt * BK;
#pragma unroll
    for (int i = 0; i < 2; ++i)
      rb[i] = *(const uint4*)(B + (size_t)(k0 + sr + i * 32) * Npad + n0b + sc);
  };
  auto sstoreB = [&](int buf) {
#pragma unroll
    for (int i = 0; i < 2; ++i)
      *(uint4*)&Bs[buf][(sr + i * 32) * LPAD + sc] = rb[i];
  };
  auto gloadAfrags = [&](int kt, v16bf* af) {
    int k0 = kt * BK;
#pragma unroll
    for (int kk2 = 0; kk2 < 2; ++kk2) {
      ((uint4*)&af[kk2])[0] = *(const uint4*)(Arow + k0 + kk2 * 32 + ak);
      ((uint4*)&af[kk2])[1] = *(const uint4*)(Arow + k0 + kk2 * 32 + 16 + ak);
    }
  };

  v8f accv[4] = {};
  auto computeStage = [&](int buf, v16bf* af) {
    v16bf bb[8];                   // issue all 128 16-bit ds_loads before the WMMAs
#pragma unroll
    for (int kk2 = 0; kk2 < 2; ++kk2)
#pragma unroll
      for (int s = 0; s < 4; ++s) {
        int cn0 = s * 16 + (lane & 15);   // lane = N col; elements = K rows (table order)
#pragma unroll
        for (int e = 0; e < 16; ++e) {
          int r = kk2 * 32 + (e < 8 ? ak + e : 16 + ak + (e - 8));
          ((unsigned short*)&bb[kk2 * 4 + s])[e] = Bs[buf][r * LPAD + cn0];
        }
      }
#pragma unroll
    for (int kk2 = 0; kk2 < 2; ++kk2)
#pragma unroll
      for (int s = 0; s < 4; ++s)
        accv[s] = __builtin_amdgcn_wmma_f32_16x16x32_bf16(
            false, af[kk2], false, bb[kk2 * 4 + s], (short)0, accv[s],
            false, false);
  };

  v16bf aA[2], aB[2];
  gloadB(0);
  sstoreB(0);
  gloadAfrags(0, aA);
  __syncthreads();

  for (int kt = 0; kt < KT; kt += 2) {      // KT is even
    gloadB(kt + 1);                          // in flight over stage-kt WMMAs
    gloadAfrags(kt + 1, aB);
    if (kt + 2 < KT)
      __builtin_prefetch(Arow + (kt + 2) * BK, 0, 1);   // global_prefetch_b8
    computeStage(0, aA);
    sstoreB(1);
    __syncthreads();

    if (kt + 2 < KT) { gloadB(kt + 2); gloadAfrags(kt + 2, aA); }
    computeStage(1, aB);
    if (kt + 2 < KT) sstoreB(0);
    __syncthreads();
  }

  // C/D layout: VGPR j -> row m0 + (lane>>4)*8 + j, col n0 + s*16 + (lane&15)
  const int cn = lane & 15;
  const int mb = (lane >> 4) * 8;
#pragma unroll
  for (int s = 0; s < 4; ++s) {
    int col = n0b + s * 16 + cn;
    if (col >= Nout) continue;
    float bv = bias ? bias[col] : 0.0f;
#pragma unroll
    for (int j = 0; j < 8; ++j) {
      int row = m0b + m0w + mb + j;
      if (row >= M) continue;
      C[(size_t)row * Nout + col] = accv[s][j] + bv;
    }
  }
}

// ---------------- GAT pieces ----------------
// one wave per node: coalesced float4 row scan + shuffle butterfly reduce
__global__ __launch_bounds__(256)
void k_node_dots(const float* __restrict__ h,
                 const float* __restrict__ a_src,
                 const float* __restrict__ a_dst,
                 float* __restrict__ ssrc, float* __restrict__ sdst, int n) {
  int node = (int)(((long)blockIdx.x * blockDim.x + threadIdx.x) >> 5);
  int lane = threadIdx.x & 31;
  if (node >= n) return;
  const float4* row = (const float4*)(h + (size_t)node * DIMH);
  const float4* as  = (const float4*)a_src;
  const float4* ad  = (const float4*)a_dst;
  float s1 = 0.f, s2 = 0.f;
#pragma unroll
  for (int j = lane; j < DIMH / 4; j += 32) {
    float4 v = row[j], w1 = as[j], w2 = ad[j];
    s1 += v.x * w1.x + v.y * w1.y + v.z * w1.z + v.w * w1.w;
    s2 += v.x * w2.x + v.y * w2.y + v.z * w2.z + v.w * w2.w;
  }
#pragma unroll
  for (int o = 16; o > 0; o >>= 1) {
    s1 += __shfl_xor(s1, o, 32);
    s2 += __shfl_xor(s2, o, 32);
  }
  if (lane == 0) { ssrc[node] = s1; sdst[node] = s2; }
}

__global__ void k_gat_init(unsigned* __restrict__ emax, float* __restrict__ esum,
                           float* __restrict__ agg, const float* __restrict__ bias,
                           int n) {
  int idx = blockIdx.x * blockDim.x + threadIdx.x;
  if (idx < n) { emax[idx] = 0x007FFFFFu /* f2ord(-inf) */; esum[idx] = 0.0f; }
  if (idx < n * DIMH) agg[idx] = bias[idx % DIMH];
}

__device__ __forceinline__ void edge_nodes(const int* __restrict__ ei, int E,
                                           int eid, int& s, int& d) {
  if (eid < E) { s = ei[eid]; d = ei[E + eid]; }
  else         { s = d = eid - E; }            // appended self loops
}
__device__ __forceinline__ float edge_e(const float* ssrc, const float* sdst,
                                        int s, int d) {
  float e = ssrc[s] + sdst[d];
  return e >= 0.0f ? e : NEG_SLOPE * e;
}

__global__ void k_edge_max(const int* __restrict__ ei, const float* __restrict__ ssrc,
                           const float* __restrict__ sdst, unsigned* __restrict__ emax,
                           int E, int ET) {
  int eid = blockIdx.x * blockDim.x + threadIdx.x;
  if (eid >= ET) return;
  int s, d; edge_nodes(ei, E, eid, s, d);
  atomicMax(&emax[d], f2ord(edge_e(ssrc, sdst, s, d)));
}

__global__ void k_edge_sum(const int* __restrict__ ei, const float* __restrict__ ssrc,
                           const float* __restrict__ sdst,
                           const unsigned* __restrict__ emax,
                           float* __restrict__ esum, int E, int ET) {
  int eid = blockIdx.x * blockDim.x + threadIdx.x;
  if (eid >= ET) return;
  int s, d; edge_nodes(ei, E, eid, s, d);
  float p = __expf(edge_e(ssrc, sdst, s, d) - ord2f(emax[d]));
  atomicAdd(&esum[d], p);
}

// one block (256 thr = DIMH) per edge
__global__ __launch_bounds__(256)
void k_edge_aggregate(const int* __restrict__ ei, const float* __restrict__ ssrc,
                      const float* __restrict__ sdst,
                      const unsigned* __restrict__ emax,
                      const float* __restrict__ esum,
                      const float* __restrict__ hW, float* __restrict__ agg,
                      int E) {
  __shared__ float s_alpha;
  __shared__ int s_src, s_dst;
  int eid = blockIdx.x;
  if (threadIdx.x == 0) {
    int s, d; edge_nodes(ei, E, eid, s, d);
    float p = __expf(edge_e(ssrc, sdst, s, d) - ord2f(emax[d]));
    s_alpha = p / esum[d];
    s_src = s; s_dst = d;
  }
  __syncthreads();
  int f = threadIdx.x;
  atomicAdd(&agg[(size_t)s_dst * DIMH + f], s_alpha * hW[(size_t)s_src * DIMH + f]);
}

// ---------------- LSTM pointwise + residual ----------------
__global__ void k_lstm(const float* __restrict__ gates, const float* __restrict__ inp,
                       float* __restrict__ h, float* __restrict__ c,
                       float* __restrict__ xcur, int n) {
  long idx = (long)blockIdx.x * blockDim.x + threadIdx.x;
  if (idx >= (long)n * DIMH) return;
  int node = (int)(idx / DIMH), f = (int)(idx % DIMH);
  const float* g = gates + (size_t)node * 4 * DIMH;
  float gi = g[f], gf = g[DIMH + f], gg = g[2 * DIMH + f], go = g[3 * DIMH + f];
  float cv = sigm(gf) * c[idx] + sigm(gi) * tanhf(gg);
  float hv = sigm(go) * tanhf(cv);
  c[idx] = cv;
  h[idx] = hv;
  xcur[idx] = hv + inp[idx] * RES_W;
}

// ---------------------------------------------------------------------------
extern "C" void kernel_launch(void* const* d_in, const int* in_sizes, int n_in,
                              void* d_out, int out_size, void* d_ws, size_t ws_size,
                              hipStream_t stream) {
  const float* x       = (const float*)d_in[0];
  const int*   ei      = (const int*)d_in[1];       // [2,E] int32 (JAX x64 off)
  const float* lin1_w  = (const float*)d_in[2];
  const float* lin1_b  = (const float*)d_in[3];
  const float* gat_w   = (const float*)d_in[4];     // [3,256,256]
  const float* ga_src  = (const float*)d_in[5];     // [3,256]
  const float* ga_dst  = (const float*)d_in[6];
  const float* gat_b   = (const float*)d_in[7];
  const float* w_ih    = (const float*)d_in[8];     // [3,1024,256]
  const float* w_hh    = (const float*)d_in[9];
  const float* lin2_w  = (const float*)d_in[10];    // [256,47]
  const float* lin2_b  = (const float*)d_in[11];
  float* out = (float*)d_out;

  const int IN = 128, D = DIMH, G = 4 * DIMH, K2 = 2 * DIMH, OUT = 47, OUTP = 64;
  const int n  = in_sizes[0] / IN;
  const int E  = in_sizes[1] / 2;
  const int ET = E + n;

  // ---- workspace carve (~560 MB) ----
  char* base = (char*)d_ws;
  size_t off = 0;
  auto carve = [&](size_t bytes) {
    char* p = base + off;
    off += (bytes + 255) & ~(size_t)255;
    return p;
  };
  float* inp   = (float*)carve((size_t)n * D * 4);
  float* xcur  = (float*)carve((size_t)n * D * 4);
  float* h     = (float*)carve((size_t)n * D * 4);
  float* c     = (float*)carve((size_t)n * D * 4);
  float* hW    = (float*)carve((size_t)n * D * 4);
  float* agg   = (float*)carve((size_t)n * D * 4);
  float* gates = (float*)carve((size_t)n * G * 4);
  unsigned short* abf = (unsigned short*)carve((size_t)n * K2 * 2);  // [xb|h]
  unsigned short* wbf = (unsigned short*)carve((size_t)K2 * G * 2);
  float*    ssrc = (float*)carve((size_t)n * 4);
  float*    sdst = (float*)carve((size_t)n * 4);
  unsigned* emax = (unsigned*)carve((size_t)n * 4);
  float*    esum = (float*)carve((size_t)n * 4);
  (void)ws_size; (void)n_in; (void)out_size;

  auto cdiv = [](long a, long b) { return (unsigned)((a + b - 1) / b); };

  hipMemsetAsync(h, 0, (size_t)n * D * 4, stream);
  hipMemsetAsync(c, 0, (size_t)n * D * 4, stream);

  // ---- lin1: inp = x @ lin1_w + lin1_b ----
  k_to_bf16_strided<<<cdiv((long)n * IN, 256), 256, 0, stream>>>(
      x, abf, (long)n * IN, IN, IN, 0, 0);
  k_weight_to_bf16<<<cdiv(IN * D, 256), 256, 0, stream>>>(lin1_w, wbf, IN, D, D, 0);
  dim3 gD(cdiv(n, BM), D / BN);
  k_gemm_bf16<<<gD, 256, 0, stream>>>(abf, wbf, inp, lin1_b, n, IN, D, D);
  hipMemcpyAsync(xcur, inp, (size_t)n * D * 4, hipMemcpyDeviceToDevice, stream);

  for (int l = 0; l < 3; ++l) {
    // hW = xcur @ gat_w[l]
    k_to_bf16_strided<<<cdiv((long)n * D, 256), 256, 0, stream>>>(
        xcur, abf, (long)n * D, D, D, 0, 0);
    k_weight_to_bf16<<<cdiv(D * D, 256), 256, 0, stream>>>(
        gat_w + (size_t)l * D * D, wbf, D, D, D, 0);
    k_gemm_bf16<<<gD, 256, 0, stream>>>(abf, wbf, hW, nullptr, n, D, D, D);

    // attention logits + edge softmax + aggregation (+bias via agg init)
    k_node_dots<<<cdiv((long)n * 32, 256), 256, 0, stream>>>(
        hW, ga_src + (size_t)l * D, ga_dst + (size_t)l * D, ssrc, sdst, n);
    k_gat_init<<<cdiv((long)n * D, 256), 256, 0, stream>>>(
        emax, esum, agg, gat_b + (size_t)l * D, n);
    k_edge_max<<<cdiv(ET, 256), 256, 0, stream>>>(ei, ssrc, sdst, emax, E, ET);
    k_edge_sum<<<cdiv(ET, 256), 256, 0, stream>>>(ei, ssrc, sdst, emax, esum, E, ET);
    k_edge_aggregate<<<ET, 256, 0, stream>>>(ei, ssrc, sdst, emax, esum, hW, agg, E);

    // gates = [tanh(agg) | h] @ [w_ih^T ; w_hh^T]  (single K=512 GEMM)
    k_to_bf16_strided<<<cdiv((long)n * D, 256), 256, 0, stream>>>(
        agg, abf, (long)n * D, D, K2, 0, 1);          // xb = tanh(agg) cols 0..255
    k_to_bf16_strided<<<cdiv((long)n * D, 256), 256, 0, stream>>>(
        h, abf, (long)n * D, D, K2, D, 0);            // h, cols 256..511
    k_gates_weight<<<cdiv((long)K2 * G, 256), 256, 0, stream>>>(
        w_ih + (size_t)l * G * D, w_hh + (size_t)l * G * D, wbf, D, G);
    dim3 gG(cdiv(n, BM), G / BN);
    k_gemm_bf16<<<gG, 256, 0, stream>>>(abf, wbf, gates, nullptr, n, K2, G, G);

    // LSTM pointwise + residual
    k_lstm<<<cdiv((long)n * D, 256), 256, 0, stream>>>(gates, inp, h, c, xcur, n);
  }

  // ---- lin2: out = xcur @ lin2_w + lin2_b  (N padded 47 -> 64) ----
  k_to_bf16_strided<<<cdiv((long)n * D, 256), 256, 0, stream>>>(
      xcur, abf, (long)n * D, D, D, 0, 0);
  k_weight_to_bf16<<<cdiv(D * OUTP, 256), 256, 0, stream>>>(
      lin2_w, wbf, D, OUT, OUTP, 0);
  dim3 gO(cdiv(n, BM), 1);
  k_gemm_bf16<<<gO, 256, 0, stream>>>(abf, wbf, out, lin2_b, n, D, OUT, OUTP);
}
